// MHABlock_48610439856167
// MI455X (gfx1250) — matrix operand order repaired and checked by reading, not compiled, not run on hardware
//
#include <hip/hip_runtime.h>
#include <hip/hip_bf16.h>

typedef _Float16 v8h  __attribute__((ext_vector_type(8)));
typedef _Float16 v16h __attribute__((ext_vector_type(16)));
typedef float    v8f  __attribute__((ext_vector_type(8)));

static __device__ __forceinline__ v16h cat8(v8h a, v8h b) {
    return __builtin_shufflevector(a, b, 0,1,2,3,4,5,6,7,8,9,10,11,12,13,14,15);
}

#define WMMA_F16(A, B, C) \
    __builtin_amdgcn_wmma_f32_16x16x32_f16(false, (A), false, (B), (short)0, (C), false, false)

// ---------------------------------------------------------------------------
// Kernel 0: convert weights f32 -> f16, folding softmax scale (0.25) into Wq.
// Wh layout: [3][128][128] (o, c) row-major, f16.
// ---------------------------------------------------------------------------
__global__ void wconv_kernel(const float* __restrict__ wq,
                             const float* __restrict__ wk,
                             const float* __restrict__ wv,
                             _Float16* __restrict__ Wh) {
    int i = blockIdx.x * 256 + threadIdx.x;          // 0 .. 49151
    if (i >= 3 * 16384) return;
    int wsel = i / 16384, r = i % 16384;
    float v = (wsel == 0) ? wq[r] : (wsel == 1) ? wk[r] : wv[r];
    if (wsel == 0) v *= 0.25f;                       // fold d^-0.5 into Q
    Wh[i] = (_Float16)v;
}

// ---------------------------------------------------------------------------
// Kernel 1: QKV projection. One wave per (b, 16-pos tile): 512 blocks x 32.
//   Qh/Kh: [b][n][pos][d] f16   (d contiguous)
//   Vt:    [b][n][d][pos] f16   (pos contiguous)
// ---------------------------------------------------------------------------
__global__ void __launch_bounds__(32)
qkv_kernel(const float* __restrict__ x, const _Float16* __restrict__ Wh,
           _Float16* __restrict__ Qh, _Float16* __restrict__ Kh,
           _Float16* __restrict__ Vt) {
    __shared__ __align__(16) _Float16 tile[16 * 136];   // (pos, c), padded row

    const int blk  = blockIdx.x;
    const int b    = blk >> 6;
    const int p0   = (blk & 63) * 16;
    const int lane = threadIdx.x;
    const int l16  = lane & 15;
    const int hi   = lane >> 4;                         // 0/1 lane half

    // ---- stage x tile (16 pos x 128 c) into LDS as f16 -------------------
    const float* xb = x + (size_t)b * 128 * 1024;
    #pragma unroll 8
    for (int it = 0; it < 64; ++it) {
        int c = it * 2 + hi;
        tile[l16 * 136 + c] = (_Float16)xb[(size_t)c * 1024 + p0 + l16];
    }
    __syncthreads();

    // ---- build x fragments (A for V path, B for Q/K path) ----------------
    v16h ax[4], bx[4];
    #pragma unroll
    for (int kc = 0; kc < 4; ++kc) {
        // A-frag: lo lanes K=c0..7,16..23 ; hi lanes +8
        const _Float16* ta = tile + l16 * 136 + kc * 32 + hi * 8;
        ax[kc] = cat8(*(const v8h*)ta, *(const v8h*)(ta + 16));
        // B-frag: lane = pos col, lo lanes K=c0..7,16..23 ; hi lanes +8
        const _Float16* tb = tile + l16 * 136 + kc * 32 + hi * 8;
        bx[kc] = cat8(*(const v8h*)tb, *(const v8h*)(tb + 16));
    }

    // ---- Q and K: A = W tile, B = x  (so d lands on VGPR axis) -----------
    #pragma unroll
    for (int wsel = 0; wsel < 2; ++wsel) {
        _Float16* dst = (wsel == 0) ? Qh : Kh;
        const _Float16* Wp = Wh + wsel * 16384;
        #pragma unroll
        for (int ot = 0; ot < 8; ++ot) {                // head index
            v8f acc = {};
            #pragma unroll
            for (int kc = 0; kc < 4; ++kc) {
                const _Float16* wp = Wp + (ot * 16 + l16) * 128 + kc * 32 + hi * 8;
                v16h aw = cat8(*(const v8h*)wp, *(const v8h*)(wp + 16));
                acc = WMMA_F16(aw, bx[kc], acc);
            }
            // C: lane -> pos = p0+l16 ; VGPR r -> d = r + 8*hi
            v8h o8;
            #pragma unroll
            for (int r = 0; r < 8; ++r) o8[r] = (_Float16)acc[r];
            _Float16* op = dst + ((size_t)(b * 8 + ot) * 1024 + p0 + l16) * 16 + hi * 8;
            *(v8h*)op = o8;
        }
    }

    // ---- V: A = x, B = W tile  (so pos lands on VGPR axis) ---------------
    #pragma unroll
    for (int ot = 0; ot < 8; ++ot) {
        v8f acc = {};
        #pragma unroll
        for (int kc = 0; kc < 4; ++kc) {
            const _Float16* wp = Wh + 2 * 16384 + (ot * 16 + l16) * 128 + kc * 32 + hi * 8;
            v16h bw = cat8(*(const v8h*)wp, *(const v8h*)(wp + 16));
            acc = WMMA_F16(ax[kc], bw, acc);
        }
        // C: lane -> d = l16 ; VGPR r -> pos = p0 + r + 8*hi
        v8h o8;
        #pragma unroll
        for (int r = 0; r < 8; ++r) o8[r] = (_Float16)acc[r];
        _Float16* op = Vt + ((size_t)(b * 8 + ot) * 16 + l16) * 1024 + p0 + hi * 8;
        *(v8h*)op = o8;
    }
}

// ---------------------------------------------------------------------------
// Kernel 2: attention. 64 blocks = (b, head), 8 waves, 8 query tiles / wave.
// Per 32-key chunk (== one softmax group over y): 2 WMMAs for S, shfl-based
// row softmax, P transposed C->A layout via per-wave LDS, 1 WMMA for O.
// ---------------------------------------------------------------------------
__global__ void __launch_bounds__(256)
attn_kernel(const _Float16* __restrict__ Qh, const _Float16* __restrict__ Kh,
            const _Float16* __restrict__ Vt, float* __restrict__ out) {
    __shared__ __align__(16) _Float16 pbuf[8 * 640];    // 8 waves x (16x40)

    const int blk  = blockIdx.x;                        // b*8 + n
    const int b    = blk >> 3;
    const int n    = blk & 7;
    const int tid  = threadIdx.x;
    const int wid  = tid >> 5;
    const int lane = tid & 31;
    const int l16  = lane & 15;
    const int hi   = lane >> 4;

    const _Float16* Qp = Qh + (size_t)blk * 1024 * 16;
    const _Float16* Kp = Kh + (size_t)blk * 1024 * 16;
    const _Float16* Vp = Vt + (size_t)blk * 16 * 1024;
    _Float16* pb = pbuf + wid * 640;
    const v8h zero8 = {};
    const float L2E = 1.44269504088896f;

    for (int i = 0; i < 8; ++i) {
        const int qt = wid * 8 + i;                     // query tile [0,64)
        // Q A-frag, K-dim d padded 16->32 (upper elements zero)
        const _Float16* qp = Qp + ((size_t)qt * 16 + l16) * 16;
        v16h aq = hi ? cat8(zero8, zero8) : cat8(*(const v8h*)qp, *(const v8h*)(qp + 8));

        v8f acc = {};
        for (int j = 0; j < 32; ++j) {                  // 32 key chunks of 32
            // K B-frags: lanes 0-15 carry d=0..15, lanes 16-31 are zero pad
            v8h k0a = {}, k0b = {}, k1a = {}, k1b = {};
            if (!hi) {
                const _Float16* kp0 = Kp + ((size_t)j * 32 + l16) * 16;
                k0a = *(const v8h*)kp0;  k0b = *(const v8h*)(kp0 + 8);
                const _Float16* kp1 = kp0 + 16 * 16;
                k1a = *(const v8h*)kp1;  k1b = *(const v8h*)(kp1 + 8);
            }
            v8f z = {};
            v8f s0 = WMMA_F16(aq, cat8(k0a, k0b), z);   // keys j*32 + 0..15
            v8f s1 = WMMA_F16(aq, cat8(k1a, k1b), z);   // keys j*32 + 16..31

            // per-row softmax over the 32 keys of this chunk
            #pragma unroll
            for (int r = 0; r < 8; ++r) {
                float t0 = s0[r], t1 = s1[r];
                float m = fmaxf(t0, t1);
                #pragma unroll
                for (int off = 1; off < 16; off <<= 1)
                    m = fmaxf(m, __shfl_xor(m, off, 32));
                float e0 = exp2f((t0 - m) * L2E);
                float e1 = exp2f((t1 - m) * L2E);
                float s = e0 + e1;
                #pragma unroll
                for (int off = 1; off < 16; off <<= 1)
                    s += __shfl_xor(s, off, 32);
                float inv = 1.0f / s;
                // C layout -> row-major P tile in LDS
                _Float16* pr = pb + (r + hi * 8) * 40;
                pr[l16]      = (_Float16)(e0 * inv);
                pr[16 + l16] = (_Float16)(e1 * inv);
            }
            asm volatile("s_wait_dscnt 0" ::: "memory"); // cross-lane LDS visibility

            // P as A-frag (K-dim = 32 keys)
            const _Float16* pr = pb + l16 * 40 + hi * 8;
            v16h ap = cat8(*(const v8h*)pr, *(const v8h*)(pr + 16));

            // V B-frag from Vt: lane = d col, keys 0..7,16..23 (hi lanes +8)
            const _Float16* vp = Vp + (size_t)l16 * 1024 + j * 32 + hi * 8;
            v16h bv = cat8(*(const v8h*)vp, *(const v8h*)(vp + 16));

            acc = WMMA_F16(ap, bv, acc);                // O += P * V
        }
        // C: lane -> channel n*16+l16 ; VGPR r -> pos qt*16 + r + 8*hi
        float* op = out + ((size_t)b * 128 + n * 16 + l16) * 1024 + qt * 16 + hi * 8;
        float4 lo4 = make_float4(acc[0], acc[1], acc[2], acc[3]);
        float4 hi4 = make_float4(acc[4], acc[5], acc[6], acc[7]);
        *(float4*)op       = lo4;
        *(float4*)(op + 4) = hi4;
    }
}

// ---------------------------------------------------------------------------
extern "C" void kernel_launch(void* const* d_in, const int* in_sizes, int n_in,
                              void* d_out, int out_size, void* d_ws, size_t ws_size,
                              hipStream_t stream) {
    const float* x  = (const float*)d_in[0];
    const float* wq = (const float*)d_in[1];
    const float* wk = (const float*)d_in[2];
    const float* wv = (const float*)d_in[3];
    float* out = (float*)d_out;

    // workspace: Qh | Kh | Vt | Wh   (f16)
    _Float16* Qh = (_Float16*)d_ws;                 // 8*8*1024*16
    _Float16* Kh = Qh + (size_t)8 * 8 * 1024 * 16;
    _Float16* Vt = Kh + (size_t)8 * 8 * 1024 * 16;
    _Float16* Wh = Vt + (size_t)8 * 8 * 1024 * 16;  // 3*128*128

    wconv_kernel<<<(3 * 16384 + 255) / 256, 256, 0, stream>>>(wq, wk, wv, Wh);
    qkv_kernel<<<8 * 64, 32, 0, stream>>>(x, Wh, Qh, Kh, Vt);
    attn_kernel<<<64, 256, 0, stream>>>(Qh, Kh, Vt, out);
}
